// OptimizedSelfAttention_78649441124527
// MI455X (gfx1250) — matrix operand order repaired and checked by reading, not compiled, not run on hardware
//
#include <hip/hip_runtime.h>

typedef __attribute__((ext_vector_type(16))) _Float16 v16h;
typedef __attribute__((ext_vector_type(8)))  _Float16 v8h;
typedef __attribute__((ext_vector_type(8)))  float    v8f;

constexpr int   C_    = 512;
constexpr int   HFULL = 256;
constexpr int   WFULL = 256;
constexpr int   HR    = 64;
constexpr int   WR    = 64;
constexpr int   N_    = 4096;     // HR*WR tokens
constexpr int   NHEAD = 8;
constexpr int   HD    = 64;
constexpr float EPS_  = 1e-5f;

#define WMMA_F16(acc, a, b)                                                     \
  (acc) = __builtin_amdgcn_wmma_f32_16x16x32_f16(false, (a), false, (b),        \
                                                 (short)0, (acc), false, false)

union AFrag { v16h v; v8h h2[2]; };

// ---------------------------------------------------------------------------
// K1: per-group partial sums (8 groups x 64ch x 65536px each)
// ---------------------------------------------------------------------------
__global__ void k_group_stats(const float* __restrict__ x, float* __restrict__ stats) {
  constexpr int SLICES = 256;
  constexpr int GSIZE  = 64 * 65536;          // elems per group
  constexpr int PERBLK = GSIZE / SLICES;      // 16384
  int g = blockIdx.x / SLICES;
  int s = blockIdx.x % SLICES;
  long long base = (long long)g * GSIZE + (long long)s * PERBLK;
  float sum = 0.f, ssq = 0.f;
  for (int i = threadIdx.x; i < PERBLK; i += 256) {
    float v = x[base + i];
    sum += v; ssq += v * v;
  }
  for (int off = 16; off; off >>= 1) {
    sum += __shfl_down(sum, off, 32);
    ssq += __shfl_down(ssq, off, 32);
  }
  __shared__ float ls[2][8];
  int wid = threadIdx.x >> 5;
  if ((threadIdx.x & 31) == 0) { ls[0][wid] = sum; ls[1][wid] = ssq; }
  __syncthreads();
  if (threadIdx.x == 0) {
    float a = 0.f, b = 0.f;
    for (int w = 0; w < 8; ++w) { a += ls[0][w]; b += ls[1][w]; }
    atomicAdd(&stats[g], a);
    atomicAdd(&stats[8 + g], b);
  }
}

// K2: finalize mean / inv-std per group
__global__ void k_finalize_stats(float* __restrict__ stats) {
  int g = threadIdx.x;
  if (g < 8) {
    const float n = 64.f * 65536.f;
    float m = stats[g] / n;
    float v = stats[8 + g] / n - m * m;
    stats[16 + g] = m;
    stats[24 + g] = rsqrtf(v + EPS_);
  }
}

// ---------------------------------------------------------------------------
// K3: 4x4 avg-pool + groupnorm affine + positional encoding -> xrT [N_][C_] f16
// (norm commutes with pooling since it's per-channel affine)
// ---------------------------------------------------------------------------
__global__ void k_pool_norm_pe(const float* __restrict__ x,
                               const float* __restrict__ gnw,
                               const float* __restrict__ gnb,
                               const float* __restrict__ stats,
                               _Float16* __restrict__ xrT) {
  int idx = blockIdx.x * 256 + threadIdx.x;          // over C_*N_ = 2M
  int c = idx & (C_ - 1);
  int n = idx >> 9;
  int hr = n >> 6, wr = n & 63;
  const float* xp = x + (long long)c * (HFULL * WFULL) + (hr * 4) * WFULL + wr * 4;
  float s = 0.f;
#pragma unroll
  for (int dy = 0; dy < 4; ++dy) {
    s += xp[dy * WFULL + 0] + xp[dy * WFULL + 1] + xp[dy * WFULL + 2] + xp[dy * WFULL + 3];
  }
  s *= (1.f / 16.f);
  int g = c >> 6;
  s = (s - stats[16 + g]) * stats[24 + g] * gnw[c] + gnb[c];
  // positional encoding: c -> (i = c>>2 freq index, j = c&3 in {sin y,cos y,sin x,cos x})
  int fi = c >> 2, j = c & 3;
  float f = __expf((float)(2 * fi) * (-9.210340371976184f / 256.f)); // 10000^{-fi/128}
  float arg = ((j < 2) ? (float)hr : (float)wr) * f;
  float pe = (j & 1) ? __cosf(arg) : __sinf(arg);
  s += 0.01f * pe;
  xrT[(long long)n * C_ + c] = (_Float16)s;
}

// K4: f32 -> f16 weight conversion (grid-stride)
__global__ void k_cvt_f16(const float* __restrict__ w, _Float16* __restrict__ o, int n) {
  for (int i = blockIdx.x * 256 + threadIdx.x; i < n; i += gridDim.x * 256)
    o[i] = (_Float16)w[i];
}

// ---------------------------------------------------------------------------
// Per-wave 16x32 WMMA slab over K=512: two adjacent 16x16 N tiles share one
// A fragment per K step (2 independent wmmas back-to-back for XDL ILP).
//   A  [M x 512] row-major f16, Bt [N x 512] row-major f16 (i.e. B^T, K-contig)
// ---------------------------------------------------------------------------
__device__ __forceinline__ void wmma_slab_k512(const _Float16* __restrict__ A,
                                               const _Float16* __restrict__ Bt,
                                               int m0, int n0,
                                               v8f& acc0, v8f& acc1) {
  int l  = threadIdx.x & 31;
  int lr = l & 15;
  int lh = l >> 4;
  const _Float16* arow  = A  + (long long)(m0 + lr) * 512;
  const _Float16* brow0 = Bt + (long long)(n0 + lr) * 512;
  const _Float16* brow1 = Bt + (long long)(n0 + 16 + lr) * 512;
  acc0 = {}; acc1 = {};
#pragma unroll 4
  for (int k0 = 0; k0 < 512; k0 += 32) {
    __builtin_prefetch(arow  + k0 + 256, 0, 3);   // WGP-scope prefetch
    __builtin_prefetch(brow0 + k0 + 256, 0, 3);
    __builtin_prefetch(brow1 + k0 + 256, 0, 3);
    AFrag a;
    a.h2[0] = *(const v8h*)(arow + k0 + 16 * lh);       // K = k0 + {0..7}/{16..23}
    a.h2[1] = *(const v8h*)(arow + k0 + 8 + 16 * lh);   // K = k0 + {8..15}/{24..31}
    v16h b0 = *(const v16h*)(brow0 + k0 + 16 * lh);     // K-contiguous per lane
    v16h b1 = *(const v16h*)(brow1 + k0 + 16 * lh);
    WMMA_F16(acc0, a.v, b0);
    WMMA_F16(acc1, a.v, b1);
  }
}

// ---------------------------------------------------------------------------
// K5: qkv GEMM (M=1536, K=512, N=4096). Epilogue scatters q/k/v in the layouts
// the flash kernel wants: qh [h][n][d] (scaled), kh [h][m][d], vh [h][d][m].
// ---------------------------------------------------------------------------
__global__ void k_qkv_gemm(const _Float16* __restrict__ Wq,
                           const _Float16* __restrict__ xrT,
                           const float* __restrict__ temperature,
                           _Float16* __restrict__ qh,
                           _Float16* __restrict__ kh,
                           _Float16* __restrict__ vh) {
  int wave = threadIdx.x >> 5;
  int m0 = (blockIdx.y * 8 + wave) * 16;
  int n0 = blockIdx.x * 32;
  v8f acc0, acc1;
  wmma_slab_k512(Wq, xrT, m0, n0, acc0, acc1);
  int l = threadIdx.x & 31, lr = l & 15, lh = l >> 4;
#pragma unroll
  for (int t = 0; t < 2; ++t) {
    int n = n0 + t * 16 + lr;
#pragma unroll
    for (int i = 0; i < 8; ++i) {
      int o = m0 + i + 8 * lh;
      float v = (t == 0) ? acc0[i] : acc1[i];
      if (o < 512) {
        int h = o >> 6, d = o & 63;
        float sc = 0.125f * temperature[h];                // hd^-0.5 * temp
        qh[((long long)h * N_ + n) * HD + d] = (_Float16)(v * sc);
      } else if (o < 1024) {
        int oo = o - 512; int h = oo >> 6, d = oo & 63;
        kh[((long long)h * N_ + n) * HD + d] = (_Float16)v;
      } else {
        int oo = o - 1024; int h = oo >> 6, d = oo & 63;
        vh[((long long)h * HD + d) * N_ + n] = (_Float16)v;
      }
    }
  }
}

// ---------------------------------------------------------------------------
// K6: flash attention. One wave owns 16 queries of one head; streams all 4096
// keys in 32-wide chunks with online softmax. P converts C-layout -> A-layout
// through a 1KB per-wave LDS buffer (LDS ops are in-order per wave).
// ---------------------------------------------------------------------------
__device__ __forceinline__ float rowmax16(float v) {
  v = fmaxf(v, __shfl_xor(v, 1, 16));
  v = fmaxf(v, __shfl_xor(v, 2, 16));
  v = fmaxf(v, __shfl_xor(v, 4, 16));
  v = fmaxf(v, __shfl_xor(v, 8, 16));
  return v;
}
__device__ __forceinline__ float rowsum16(float v) {
  v += __shfl_xor(v, 1, 16);
  v += __shfl_xor(v, 2, 16);
  v += __shfl_xor(v, 4, 16);
  v += __shfl_xor(v, 8, 16);
  return v;
}

__global__ void k_flash_attn(const _Float16* __restrict__ qh,
                             const _Float16* __restrict__ kh,
                             const _Float16* __restrict__ vh,
                             _Float16* __restrict__ attnT) {
  __shared__ _Float16 plds[4][16 * 32];
  int wave = threadIdx.x >> 5;
  int gw = blockIdx.x * 4 + wave;          // 2048 waves = 8 heads * 256 qtiles
  int h  = gw >> 8;
  int qt = gw & 255;
  int l = threadIdx.x & 31, lr = l & 15, lh = l >> 4;

  const _Float16* Q = qh + (long long)h * N_ * HD;
  const _Float16* K = kh + (long long)h * N_ * HD;
  const _Float16* V = vh + (long long)h * HD * N_;
  _Float16* P = plds[wave];

  // Q A-fragments for d in [0,32) and [32,64)
  const _Float16* qrow = Q + (long long)(qt * 16 + lr) * HD;
  AFrag aq0, aq1;
  aq0.h2[0] = *(const v8h*)(qrow +  0 + 16 * lh);
  aq0.h2[1] = *(const v8h*)(qrow +  8 + 16 * lh);
  aq1.h2[0] = *(const v8h*)(qrow + 32 + 16 * lh);
  aq1.h2[1] = *(const v8h*)(qrow + 40 + 16 * lh);

  float rmax[8], rsum[8];
  v8f o0 = {}, o1 = {}, o2 = {}, o3 = {};
#pragma unroll
  for (int i = 0; i < 8; ++i) { rmax[i] = -1e30f; rsum[i] = 0.f; }

  for (int m0 = 0; m0 < N_; m0 += 32) {
    // run-ahead prefetch of next chunk's K and V lines (WGP scope)
    __builtin_prefetch(K + (long long)(m0 + 32 + lr) * HD + 16 * lh, 0, 3);
    __builtin_prefetch(K + (long long)(m0 + 48 + lr) * HD + 16 * lh, 0, 3);
    __builtin_prefetch(V + (long long)( 0 + lr) * N_ + m0 + 32 + 16 * lh, 0, 3);
    __builtin_prefetch(V + (long long)(16 + lr) * N_ + m0 + 32 + 16 * lh, 0, 3);
    __builtin_prefetch(V + (long long)(32 + lr) * N_ + m0 + 32 + 16 * lh, 0, 3);
    __builtin_prefetch(V + (long long)(48 + lr) * N_ + m0 + 32 + 16 * lh, 0, 3);
    // --- S = Q K^T for 32 keys (two 16-col tiles) ---
    v8f s0 = {}, s1 = {};
    {
      const _Float16* krow = K + (long long)(m0 + lr) * HD;
      v16h b0 = *(const v16h*)(krow + 16 * lh);
      v16h b1 = *(const v16h*)(krow + 32 + 16 * lh);
      WMMA_F16(s0, aq0.v, b0);
      WMMA_F16(s0, aq1.v, b1);
    }
    {
      const _Float16* krow = K + (long long)(m0 + 16 + lr) * HD;
      v16h b0 = *(const v16h*)(krow + 16 * lh);
      v16h b1 = *(const v16h*)(krow + 32 + 16 * lh);
      WMMA_F16(s1, aq0.v, b0);
      WMMA_F16(s1, aq1.v, b1);
    }
    // --- online softmax update per row (VGPR i holds row i + 8*lh) ---
#pragma unroll
    for (int i = 0; i < 8; ++i) {
      float mx = rowmax16(fmaxf(s0[i], s1[i]));
      float mnew = fmaxf(rmax[i], mx);
      float sc = __expf(rmax[i] - mnew);
      rmax[i] = mnew;
      float p0 = __expf(s0[i] - mnew);
      float p1 = __expf(s1[i] - mnew);
      rsum[i] = rsum[i] * sc + rowsum16(p0 + p1);
      o0[i] *= sc; o1[i] *= sc; o2[i] *= sc; o3[i] *= sc;
      int row = i + 8 * lh;
      P[row * 32 + lr]      = (_Float16)p0;
      P[row * 32 + 16 + lr] = (_Float16)p1;
    }
    asm volatile("s_wait_dscnt 0" ::: "memory");
    // --- reload P as A-layout 16x32 fragment ---
    AFrag ap;
    {
      const _Float16* prow = P + lr * 32;
      ap.h2[0] = *(const v8h*)(prow + 16 * lh);
      ap.h2[1] = *(const v8h*)(prow + 8 + 16 * lh);
    }
    // --- O += P @ V (4 d-tiles of 16) ---
    {
      v16h bv;
      bv = *(const v16h*)(V + (long long)( 0 + lr) * N_ + m0 + 16 * lh); WMMA_F16(o0, ap.v, bv);
      bv = *(const v16h*)(V + (long long)(16 + lr) * N_ + m0 + 16 * lh); WMMA_F16(o1, ap.v, bv);
      bv = *(const v16h*)(V + (long long)(32 + lr) * N_ + m0 + 16 * lh); WMMA_F16(o2, ap.v, bv);
      bv = *(const v16h*)(V + (long long)(48 + lr) * N_ + m0 + 16 * lh); WMMA_F16(o3, ap.v, bv);
    }
  }
  // --- normalize and scatter: attnT[n][h*64 + d] ---
#pragma unroll
  for (int i = 0; i < 8; ++i) {
    float inv = 1.f / rsum[i];
    long long n = qt * 16 + i + 8 * lh;
    long long base = n * C_ + h * HD + lr;
    attnT[base +  0] = (_Float16)(o0[i] * inv);
    attnT[base + 16] = (_Float16)(o1[i] * inv);
    attnT[base + 32] = (_Float16)(o2[i] * inv);
    attnT[base + 48] = (_Float16)(o3[i] * inv);
  }
}

// ---------------------------------------------------------------------------
// K7: proj GEMM at LOW resolution (proj commutes with bilinear resize).
// proj_lo[o][n] = proj_w[o][:] . attnT[n][:]
// ---------------------------------------------------------------------------
__global__ void k_proj_gemm(const _Float16* __restrict__ Wp,
                            const _Float16* __restrict__ attnT,
                            float* __restrict__ proj_lo) {
  int wave = threadIdx.x >> 5;
  int m0 = (blockIdx.y * 8 + wave) * 16;
  int n0 = blockIdx.x * 32;
  v8f acc0, acc1;
  wmma_slab_k512(Wp, attnT, m0, n0, acc0, acc1);
  int l = threadIdx.x & 31, lr = l & 15, lh = l >> 4;
#pragma unroll
  for (int i = 0; i < 8; ++i) {
    int o = m0 + i + 8 * lh;
    proj_lo[(long long)o * N_ + n0 + lr]      = acc0[i];
    proj_lo[(long long)o * N_ + n0 + 16 + lr] = acc1[i];
  }
}

// ---------------------------------------------------------------------------
// K8: final = identity + gamma * (bilerp_x4(proj_lo) + proj_b)
// half-pixel bilinear, edge-clamped taps (matches jax.image.resize upsample)
// ---------------------------------------------------------------------------
__global__ void k_final(const float* __restrict__ x,
                        const float* __restrict__ proj_lo,
                        const float* __restrict__ proj_b,
                        const float* __restrict__ gamma,
                        float* __restrict__ out) {
  long long idx = (long long)blockIdx.x * 256 + threadIdx.x; // C_*65536 total
  int c  = (int)(idx >> 16);
  int p  = (int)(idx & 65535);
  int y  = p >> 8, xx = p & 255;
  float sy = (y  + 0.5f) * 0.25f - 0.5f;
  float sx = (xx + 0.5f) * 0.25f - 0.5f;
  float fy0 = floorf(sy), fx0 = floorf(sx);
  float wy = sy - fy0, wx = sx - fx0;
  int y0 = max((int)fy0, 0), y1 = min((int)fy0 + 1, HR - 1);
  int x0 = max((int)fx0, 0), x1 = min((int)fx0 + 1, WR - 1);
  const float* pl = proj_lo + (long long)c * N_;
  float v00 = pl[y0 * WR + x0], v01 = pl[y0 * WR + x1];
  float v10 = pl[y1 * WR + x0], v11 = pl[y1 * WR + x1];
  float bil = (1.f - wy) * ((1.f - wx) * v00 + wx * v01)
            +        wy  * ((1.f - wx) * v10 + wx * v11);
  out[idx] = x[idx] + gamma[0] * (bil + proj_b[c]);
}

// ---------------------------------------------------------------------------
extern "C" void kernel_launch(void* const* d_in, const int* in_sizes, int n_in,
                              void* d_out, int out_size, void* d_ws, size_t ws_size,
                              hipStream_t stream) {
  const float* x     = (const float*)d_in[0];
  const float* gnw   = (const float*)d_in[1];
  const float* gnb   = (const float*)d_in[2];
  const float* qkvw  = (const float*)d_in[3];
  const float* projw = (const float*)d_in[4];
  const float* projb = (const float*)d_in[5];
  const float* gamma = (const float*)d_in[6];
  const float* temp  = (const float*)d_in[7];
  float* out = (float*)d_out;

  char* ws = (char*)d_ws;
  size_t off = 0;
  float*     stats   = (float*)(ws + off);      off += 256;
  _Float16*  wq16    = (_Float16*)(ws + off);   off += (size_t)1536 * 512 * 2;
  _Float16*  wp16    = (_Float16*)(ws + off);   off += (size_t)512 * 512 * 2;
  _Float16*  xrT     = (_Float16*)(ws + off);   off += (size_t)N_ * C_ * 2;
  _Float16*  qh      = (_Float16*)(ws + off);   off += (size_t)NHEAD * N_ * HD * 2;
  _Float16*  kh      = (_Float16*)(ws + off);   off += (size_t)NHEAD * N_ * HD * 2;
  _Float16*  vh      = (_Float16*)(ws + off);   off += (size_t)NHEAD * N_ * HD * 2;
  _Float16*  attnT   = (_Float16*)(ws + off);   off += (size_t)N_ * C_ * 2;
  float*     proj_lo = (float*)(ws + off);      off += (size_t)C_ * N_ * 4;

  hipMemsetAsync(stats, 0, 256, stream);

  k_group_stats   <<<8 * 256, 256, 0, stream>>>(x, stats);
  k_finalize_stats<<<1, 32, 0, stream>>>(stats);
  k_pool_norm_pe  <<<(C_ * N_) / 256, 256, 0, stream>>>(x, gnw, gnb, stats, xrT);
  k_cvt_f16       <<<512, 256, 0, stream>>>(qkvw, wq16, 1536 * 512);
  k_cvt_f16       <<<512, 256, 0, stream>>>(projw, wp16, 512 * 512);
  k_qkv_gemm      <<<dim3(128, 12), 256, 0, stream>>>(wq16, xrT, temp, qh, kh, vh);
  k_flash_attn    <<<512, 128, 0, stream>>>(qh, kh, vh, attnT);
  k_proj_gemm     <<<dim3(128, 4), 256, 0, stream>>>(wp16, attnT, proj_lo);
  k_final         <<<(C_ * HFULL * WFULL) / 256, 256, 0, stream>>>(x, proj_lo, projb, gamma, out);
}
